// ReducedRelPositionMultiHeadedAttention_46359876993609
// MI455X (gfx1250) — matrix operand order repaired and checked by reading, not compiled, not run on hardware
//
#include <hip/hip_runtime.h>
#include <hip/hip_bf16.h>

typedef __bf16 bf16_t;
typedef __attribute__((ext_vector_type(16))) __bf16 v16bf;
typedef __attribute__((ext_vector_type(8)))  __bf16 v8bf;
typedef __attribute__((ext_vector_type(8)))  float  v8f;

#define WMMA_BF16(A_, B_, C_) \
  __builtin_amdgcn_wmma_f32_16x16x32_bf16(false, (A_), false, (B_), (short)0, (C_), false, false)

static constexpr int Tn = 1024;  // sequence length
static constexpr int Bn = 4;     // batch
static constexpr int Cn = 1024;  // channels
static constexpr int Hn = 16;    // heads
static constexpr int Dn = 64;    // head dim

// ---------------- WMMA fragment loaders (bf16 16x16x32) ----------------
// A (16xK=32, MxK): lane l -> row (l&15); K chunks {h..h+7, 16+h..16+h+7}, h=8*(l>=16)
__device__ __forceinline__ v16bf load_a_frag(const bf16_t* base, int ld, int row0, int k0, int lane) {
  const int m    = row0 + (lane & 15);
  const int half = (lane >> 4) << 3;
  const bf16_t* p = base + (size_t)m * ld + k0 + half;
  v8bf c0 = *reinterpret_cast<const v8bf*>(p);
  v8bf c1 = *reinterpret_cast<const v8bf*>(p + 16);
  v16bf r;
#pragma unroll
  for (int i = 0; i < 8; ++i) { r[i] = c0[i]; r[i + 8] = c1[i]; }
  return r;
}
// B (K=32 x 16, from row-major Bsrc[N rows, K]): lane l -> col (l&15); K chunks {h..h+7, 16+h..16+h+7}, h=8*(l>=16)
__device__ __forceinline__ v16bf load_b_frag(const bf16_t* base, int ld, int row0, int k0, int lane) {
  const int n    = row0 + (lane & 15);
  const int koff = k0 + ((lane >> 4) << 3);
  const bf16_t* p = base + (size_t)n * ld + koff;
  v8bf c0 = *reinterpret_cast<const v8bf*>(p);
  v8bf c1 = *reinterpret_cast<const v8bf*>(p + 16);
  v16bf r;
#pragma unroll
  for (int i = 0; i < 8; ++i) { r[i] = c0[i]; r[i + 8] = c1[i]; }
  return r;
}

// ---------------- conversion kernels ----------------
__global__ void cvt_tbc_btc(const float* __restrict__ in, bf16_t* __restrict__ out, int n) {
  int i = blockIdx.x * 256 + threadIdx.x;
  if (i >= n) return;
  int t = i >> 12;               // /(B*C)
  int b = (i >> 10) & 3;
  int c = i & 1023;
  out[(((size_t)b * Tn + t) << 10) + c] = (bf16_t)in[i];
}
__global__ void cvt_plain(const float* __restrict__ in, bf16_t* __restrict__ out, int n) {
  int i = blockIdx.x * 256 + threadIdx.x;
  if (i < n) out[i] = (bf16_t)in[i];
}
__global__ void cvt_pos(const float* __restrict__ in, bf16_t* __restrict__ out, int n) {
  int i = blockIdx.x * 256 + threadIdx.x;   // n = 2048*1024, source has 2047 rows
  if (i >= n) return;
  int r  = i >> 10;
  int rs = r > 2046 ? 2046 : r;
  out[i] = (bf16_t)in[((size_t)rs << 10) + (i & 1023)];
}

// ---------------- generic bf16 WMMA GEMM: Y = A @ Bsrc^T (+bias) ----------------
// MODE 0: bf16 out (optional bias)                 -> Y[coff + m*ldc + n]
// MODE 1: dual out: qu = acc+bias+bias2, qv = acc+bias+bias3 (bf16)
// MODE 2: bf16 out, V-transposed: [b,h,d,s] layout
// MODE 3: f32 out, [T,B,C]-transposed, +bias
template<int MODE>
__global__ void __launch_bounds__(128)
gemm_bf16_k(const bf16_t* __restrict__ A, const bf16_t* __restrict__ Bm,
            void* __restrict__ Y, bf16_t* __restrict__ Y2,
            const float* __restrict__ bias, const float* __restrict__ bias2,
            const float* __restrict__ bias3,
            int lda, int ldb, int ldc, int K,
            int zdiv, long long sA1, long long sA2,
            long long sB1, long long sB2, long long sC1, long long sC2)
{
  const int lane = threadIdx.x;
  const int z  = blockIdx.z;
  const int zq = z / zdiv, zr = z % zdiv;
  const bf16_t* Ab = A  + zq * sA1 + zr * sA2;
  const bf16_t* Bb = Bm + zq * sB1 + zr * sB2;
  const long long coff = zq * sC1 + zr * sC2;
  const int m0 = (blockIdx.y * 4 + threadIdx.y) * 64;
  const int n0 = blockIdx.x * 64;

  v8f acc[4][4] = {};
  for (int k0 = 0; k0 < K; k0 += 32) {
    v16bf af[4], bf[4];
#pragma unroll
    for (int i = 0; i < 4; ++i) af[i] = load_a_frag(Ab, lda, m0 + 16 * i, k0, lane);
#pragma unroll
    for (int j = 0; j < 4; ++j) bf[j] = load_b_frag(Bb, ldb, n0 + 16 * j, k0, lane);
#pragma unroll
    for (int i = 0; i < 4; ++i)
#pragma unroll
      for (int j = 0; j < 4; ++j)
        acc[i][j] = WMMA_BF16(af[i], bf[j], acc[i][j]);
  }

  const int mh = (lane >> 4) << 3;
  const int nl = lane & 15;
#pragma unroll
  for (int i = 0; i < 4; ++i) {
#pragma unroll
    for (int j = 0; j < 4; ++j) {
#pragma unroll
      for (int r = 0; r < 8; ++r) {
        int m = m0 + 16 * i + mh + r;
        int n = n0 + 16 * j + nl;
        float v = acc[i][j][r];
        if (MODE == 0) {
          if (bias) v += bias[n];
          ((bf16_t*)Y)[coff + (long long)m * ldc + n] = (bf16_t)v;
        } else if (MODE == 1) {
          v += bias[n];
          ((bf16_t*)Y)[(long long)m * ldc + n] = (bf16_t)(v + bias2[n]);
          Y2[(long long)m * ldc + n]           = (bf16_t)(v + bias3[n]);
        } else if (MODE == 2) {
          v += bias[n];
          int bb = m >> 10, s = m & 1023, hh = n >> 6, d = n & 63;
          ((bf16_t*)Y)[(((long long)((bb * Hn + hh) * Dn + d)) << 10) + s] = (bf16_t)v;
        } else {
          v += bias[n];
          int bb = m >> 10, t = m & 1023;
          ((float*)Y)[((long long)(t * Bn + bb) << 10) + n] = v;
        }
      }
    }
  }
}

// ---------------- fused scores + rel_shift + softmax + attn@V ----------------
// One workgroup = one (b,h) pair x 32 query rows. 8 waves, dynamic LDS:
//   sc[32][1024] f32 (scores), bd[32][1056] f32 (bd window, reused as bf16 probs)
__global__ void __launch_bounds__(256)
attn_fused_kernel(const bf16_t* __restrict__ qu, const bf16_t* __restrict__ qv,
                  const bf16_t* __restrict__ kp, const bf16_t* __restrict__ pp,
                  const bf16_t* __restrict__ vT, bf16_t* __restrict__ xout)
{
  extern __shared__ float lds[];
  float*  sc    = lds;                          // 32*1024 f32
  float*  bd    = lds + 32 * 1024;              // 32*1056 f32
  bf16_t* probs = (bf16_t*)(lds + 32 * 1024);   // 32*1024 bf16 (reuses bd space)

  const int lane = threadIdx.x;
  const int w    = threadIdx.y;
  const int t0   = blockIdx.x * 32;
  const int z    = blockIdx.y;                  // z = b*H + h
  const int b    = z >> 4;
  const int h    = z & 15;
  const int mh   = (lane >> 4) << 3;
  const int nl   = lane & 15;

  const size_t qoff = ((size_t)(b * Tn + t0)) * Cn + h * Dn;

  // ---- phase 1: bd window = (q + pos_bias_v) @ p^T,  p rows [T-32-t0, +1056) ----
  v16bf aq[2][2];
#pragma unroll
  for (int rt = 0; rt < 2; ++rt)
#pragma unroll
    for (int kc = 0; kc < 2; ++kc)
      aq[rt][kc] = load_a_frag(qv + qoff, Cn, rt * 16, kc * 32, lane);

  const int p_base = Tn - 32 - t0;
  const bf16_t* pb = pp + (size_t)p_base * Cn + h * Dn;
  // 66 column tiles = 33 pairs; 4 independent accumulators per iteration
  for (int cp = w; cp < 33; cp += 8) {
    const int ct0 = cp * 2;
    v16bf b00 = load_b_frag(pb, Cn, ct0 * 16,        0,  lane);
    v16bf b01 = load_b_frag(pb, Cn, ct0 * 16,        32, lane);
    v16bf b10 = load_b_frag(pb, Cn, (ct0 + 1) * 16,  0,  lane);
    v16bf b11 = load_b_frag(pb, Cn, (ct0 + 1) * 16,  32, lane);
    v8f acc[2][2] = {};
    acc[0][0] = WMMA_BF16(aq[0][0], b00, acc[0][0]);
    acc[0][1] = WMMA_BF16(aq[0][0], b10, acc[0][1]);
    acc[1][0] = WMMA_BF16(aq[1][0], b00, acc[1][0]);
    acc[1][1] = WMMA_BF16(aq[1][0], b10, acc[1][1]);
    acc[0][0] = WMMA_BF16(aq[0][1], b01, acc[0][0]);
    acc[0][1] = WMMA_BF16(aq[0][1], b11, acc[0][1]);
    acc[1][0] = WMMA_BF16(aq[1][1], b01, acc[1][0]);
    acc[1][1] = WMMA_BF16(aq[1][1], b11, acc[1][1]);
#pragma unroll
    for (int rt = 0; rt < 2; ++rt)
#pragma unroll
      for (int cc = 0; cc < 2; ++cc)
#pragma unroll
        for (int r = 0; r < 8; ++r)
          bd[(rt * 16 + mh + r) * 1056 + (ct0 + cc) * 16 + nl] = acc[rt][cc][r];
  }
  __syncthreads();

  // ---- phase 2: ac = (q + pos_bias_u) @ k^T ; add shifted bd, scale, clip ----
#pragma unroll
  for (int rt = 0; rt < 2; ++rt)
#pragma unroll
    for (int kc = 0; kc < 2; ++kc)
      aq[rt][kc] = load_a_frag(qu + qoff, Cn, rt * 16, kc * 32, lane);

  const bf16_t* kb = kp + (size_t)b * Tn * Cn + h * Dn;
  // 64 column tiles = 32 pairs
  for (int cp = w; cp < 32; cp += 8) {
    const int ct0 = cp * 2;
    v16bf b00 = load_b_frag(kb, Cn, ct0 * 16,        0,  lane);
    v16bf b01 = load_b_frag(kb, Cn, ct0 * 16,        32, lane);
    v16bf b10 = load_b_frag(kb, Cn, (ct0 + 1) * 16,  0,  lane);
    v16bf b11 = load_b_frag(kb, Cn, (ct0 + 1) * 16,  32, lane);
    v8f acc[2][2] = {};
    acc[0][0] = WMMA_BF16(aq[0][0], b00, acc[0][0]);
    acc[0][1] = WMMA_BF16(aq[0][0], b10, acc[0][1]);
    acc[1][0] = WMMA_BF16(aq[1][0], b00, acc[1][0]);
    acc[1][1] = WMMA_BF16(aq[1][0], b10, acc[1][1]);
    acc[0][0] = WMMA_BF16(aq[0][1], b01, acc[0][0]);
    acc[0][1] = WMMA_BF16(aq[0][1], b11, acc[0][1]);
    acc[1][0] = WMMA_BF16(aq[1][1], b01, acc[1][0]);
    acc[1][1] = WMMA_BF16(aq[1][1], b11, acc[1][1]);
#pragma unroll
    for (int rt = 0; rt < 2; ++rt) {
#pragma unroll
      for (int cc = 0; cc < 2; ++cc) {
#pragma unroll
        for (int r = 0; r < 8; ++r) {
          int m = rt * 16 + mh + r;
          int s = (ct0 + cc) * 16 + nl;
          float val = (acc[rt][cc][r] + bd[m * 1056 + s + 31 - m]) * 0.125f;  // 1/sqrt(64)
          val = fminf(fmaxf(val, -1e8f), 1e8f);
          sc[m * 1024 + s] = val;
        }
      }
    }
  }
  __syncthreads();

  // ---- phase 3: row softmax (wave32 shuffle reduce), bf16 probs -> LDS ----
#pragma unroll 1
  for (int rr = 0; rr < 4; ++rr) {
    int m = w * 4 + rr;
    const float* row = sc + m * 1024;
    float vals[32];
    float mx = -3.0e38f;
#pragma unroll
    for (int j = 0; j < 32; ++j) { vals[j] = row[lane + j * 32]; mx = fmaxf(mx, vals[j]); }
#pragma unroll
    for (int msk = 16; msk > 0; msk >>= 1) mx = fmaxf(mx, __shfl_xor(mx, msk, 32));
    float sum = 0.f;
#pragma unroll
    for (int j = 0; j < 32; ++j) { vals[j] = __expf(vals[j] - mx); sum += vals[j]; }
#pragma unroll
    for (int msk = 16; msk > 0; msk >>= 1) sum += __shfl_xor(sum, msk, 32);
    float inv = 1.f / sum;
    bf16_t* prow = probs + m * 1024;
#pragma unroll
    for (int j = 0; j < 32; ++j) prow[lane + j * 32] = (bf16_t)(vals[j] * inv);
  }
  __syncthreads();

  // ---- phase 4: x[32,64] = probs @ v  (B from vT[b,h,d,s]); split-K, 2 chains ----
  const int rt = w >> 2;   // 0..1 (row tile)
  const int nt = w & 3;    // 0..3 (d tile)
  const bf16_t* vb = vT + (size_t)z * Dn * Tn;
  v8f acc0 = {}, acc1 = {};
  for (int k0 = 0; k0 < Tn; k0 += 64) {
    v16bf a0  = load_a_frag(probs, 1024, rt * 16, k0,      lane);   // LDS reads
    v16bf bb0 = load_b_frag(vb,    Tn,   nt * 16, k0,      lane);
    v16bf a1  = load_a_frag(probs, 1024, rt * 16, k0 + 32, lane);
    v16bf bb1 = load_b_frag(vb,    Tn,   nt * 16, k0 + 32, lane);
    acc0 = WMMA_BF16(a0, bb0, acc0);
    acc1 = WMMA_BF16(a1, bb1, acc1);
  }
  const size_t obase = ((size_t)(b * Tn + t0)) * Cn + h * Dn;
#pragma unroll
  for (int r = 0; r < 8; ++r) {
    int m = rt * 16 + mh + r;
    int d = nt * 16 + nl;
    xout[obase + (size_t)m * Cn + d] = (bf16_t)(acc0[r] + acc1[r]);
  }
}

// ---------------- launcher ----------------
extern "C" void kernel_launch(void* const* d_in, const int* in_sizes, int n_in,
                              void* d_out, int out_size, void* d_ws, size_t ws_size,
                              hipStream_t stream) {
  (void)in_sizes; (void)n_in; (void)out_size; (void)ws_size;
  const float* query = (const float*)d_in[0];
  const float* key   = (const float*)d_in[1];
  const float* value = (const float*)d_in[2];
  const float* pos   = (const float*)d_in[3];
  const float* Wq    = (const float*)d_in[4];
  const float* bq    = (const float*)d_in[5];
  const float* Wk    = (const float*)d_in[6];
  const float* bk    = (const float*)d_in[7];
  const float* Wv    = (const float*)d_in[8];
  const float* bv    = (const float*)d_in[9];
  const float* Wp    = (const float*)d_in[10];
  const float* Wo    = (const float*)d_in[11];
  const float* bo    = (const float*)d_in[12];
  const float* pbu   = (const float*)d_in[13];
  const float* pbv   = (const float*)d_in[14];

  char* ws = (char*)d_ws;
  size_t off = 0;
  auto alloc = [&](size_t elems) {
    bf16_t* p = (bf16_t*)(ws + off);
    off += ((elems * sizeof(bf16_t) + 255) / 256) * 256;
    return p;
  };
  const size_t MC = 4096 * 1024;   // [B*T, C]
  bf16_t* qin = alloc(MC);
  bf16_t* kin = alloc(MC);
  bf16_t* vin = alloc(MC);
  bf16_t* pin = alloc(2048 * 1024);
  bf16_t* wq  = alloc(1024 * 1024);
  bf16_t* wk  = alloc(1024 * 1024);
  bf16_t* wv  = alloc(1024 * 1024);
  bf16_t* wp  = alloc(1024 * 1024);
  bf16_t* wo  = alloc(1024 * 1024);
  bf16_t* qu  = alloc(MC);
  bf16_t* qvb = alloc(MC);
  bf16_t* kp  = alloc(MC);
  bf16_t* vT  = alloc(MC);          // [B,H,D,T]
  bf16_t* pp  = alloc(2048 * 1024);
  bf16_t* xb  = alloc(MC);

  // 1) precision conversion (with [T,B,C] -> [B,T,C] swap for q/k/v)
  cvt_tbc_btc<<<16384, 256, 0, stream>>>(query, qin, 4194304);
  cvt_tbc_btc<<<16384, 256, 0, stream>>>(key,   kin, 4194304);
  cvt_tbc_btc<<<16384, 256, 0, stream>>>(value, vin, 4194304);
  cvt_pos    <<<8192,  256, 0, stream>>>(pos,   pin, 2097152);
  cvt_plain  <<<4096,  256, 0, stream>>>(Wq, wq, 1048576);
  cvt_plain  <<<4096,  256, 0, stream>>>(Wk, wk, 1048576);
  cvt_plain  <<<4096,  256, 0, stream>>>(Wv, wv, 1048576);
  cvt_plain  <<<4096,  256, 0, stream>>>(Wp, wp, 1048576);
  cvt_plain  <<<4096,  256, 0, stream>>>(Wo, wo, 1048576);

  // 2) projections (bf16 WMMA GEMMs, f32 accumulate)
  dim3 gblk(32, 4);
  gemm_bf16_k<1><<<dim3(16, 16, 1), gblk, 0, stream>>>(qin, wq, qu, qvb, bq, pbu, pbv,
      1024, 1024, 1024, 1024, 1, 0, 0, 0, 0, 0, 0);
  gemm_bf16_k<0><<<dim3(16, 16, 1), gblk, 0, stream>>>(kin, wk, kp, nullptr, bk, nullptr, nullptr,
      1024, 1024, 1024, 1024, 1, 0, 0, 0, 0, 0, 0);
  gemm_bf16_k<2><<<dim3(16, 16, 1), gblk, 0, stream>>>(vin, wv, vT, nullptr, bv, nullptr, nullptr,
      1024, 1024, 1024, 1024, 1, 0, 0, 0, 0, 0, 0);
  gemm_bf16_k<0><<<dim3(16, 8, 1), gblk, 0, stream>>>(pin, wp, pp, nullptr, nullptr, nullptr, nullptr,
      1024, 1024, 1024, 1024, 1, 0, 0, 0, 0, 0, 0);

  // 3) fused scores + rel_shift + softmax + attn@V (probs stay in LDS)
  attn_fused_kernel<<<dim3(32, 64), dim3(32, 8), 266240, stream>>>(qu, qvb, kp, pp, vT, xb);

  // 4) out = x @ Wo^T + bo, written as [T,B,C] f32
  gemm_bf16_k<3><<<dim3(16, 16, 1), gblk, 0, stream>>>(xb, wo, d_out, nullptr, bo, nullptr, nullptr,
      1024, 1024, 1024, 1024, 1, 0, 0, 0, 0, 0, 0);
}
